// WavePolyConv_67628555043189
// MI455X (gfx1250) — matrix operand order repaired and checked by reading, not compiled, not run on hardware
//
#include <hip/hip_runtime.h>
#include <hip/hip_bf16.h>

typedef __attribute__((ext_vector_type(16))) _Float16 v16h;
typedef __attribute__((ext_vector_type(8)))  _Float16 v8h;
typedef __attribute__((ext_vector_type(8)))  float    v8f;

#define CHN 64          // channels (C)
#define DTDT 1.0f       // DT*DT with DT=1.0
#define WROW 72         // padded halves per W row in LDS (144B: 16B-aligned, bank spread 36*m mod 64)
#define TROW 68         // padded floats per acc-tile row in LDS (272B: 16B-aligned)

// ---------------------------------------------------------------------------
// deg[i] = 1.0 (self loop)  -- init pass, d_ws is poisoned each run
// ---------------------------------------------------------------------------
__global__ void deg_init_kernel(float* __restrict__ deg, int n) {
    int i = blockIdx.x * blockDim.x + threadIdx.x;
    if (i < n) deg[i] = 1.0f;
}

// deg[dst[e]] += 1 over E directed edges
__global__ void deg_count_kernel(const int* __restrict__ dst,
                                 float* __restrict__ deg, int e) {
    int i = blockIdx.x * blockDim.x + threadIdx.x;
    if (i < e) atomicAdd(&deg[dst[i]], 1.0f);
}

// deg -> dinv = rsqrt(deg) in place (deg >= 1 always, keep guard anyway)
__global__ void dinv_kernel(float* __restrict__ deg, int n) {
    int i = blockIdx.x * blockDim.x + threadIdx.x;
    if (i < n) {
        float d = deg[i];
        deg[i] = (d > 0.0f) ? rsqrtf(d) : 0.0f;
    }
}

// ---------------------------------------------------------------------------
// Hop init: hdst[i,:] = hsrc[i,:] * dinv[i]^2   (self-loop term; every element
// written, so no separate zeroing). One thread per float4 group.
// ---------------------------------------------------------------------------
__global__ void hop_init_kernel(const float* __restrict__ hsrc,
                                const float* __restrict__ dinv,
                                float* __restrict__ hdst, int ngroups) {
    int i = blockIdx.x * blockDim.x + threadIdx.x;
    if (i >= ngroups) return;
    int node = i >> 4;                       // 16 float4 groups per node
    float di = dinv[node];
    float s = di * di;
    float4 v = ((const float4*)hsrc)[i];
    v.x *= s; v.y *= s; v.z *= s; v.w *= s;
    ((float4*)hdst)[i] = v;
}

// ---------------------------------------------------------------------------
// Edge scatter: one wave (32 lanes) per edge, each lane owns 2 channels.
// hdst[d, c] += hsrc[s, c] * dinv[s]*dinv[d]   via global_atomic_add_f32.
// ---------------------------------------------------------------------------
__global__ void edge_scatter_kernel(const float* __restrict__ hsrc,
                                    const float* __restrict__ dinv,
                                    const int* __restrict__ src,
                                    const int* __restrict__ dst,
                                    float* __restrict__ hdst, int e) {
    int t = blockIdx.x * blockDim.x + threadIdx.x;
    int edge = t >> 5;
    if (edge >= e) return;
    int lane = t & 31;
    int s = src[edge];
    int d = dst[edge];
    float nrm = dinv[s] * dinv[d];
    float2 v = ((const float2*)(hsrc + (size_t)s * CHN))[lane];
    float* o = hdst + (size_t)d * CHN + lane * 2;
    atomicAdd(o + 0, v.x * nrm);
    atomicAdd(o + 1, v.y * nrm);
}

// ---------------------------------------------------------------------------
// Fused final:  out = 2*x + DT^2 * (h @ W^T + b) - x_pre
// One wave computes a 16(node) x 64(channel) strip: 4 col-tiles of 16,
// each col-tile = 2 chained v_wmma_f32_16x16x32_f16 (K=64).
// W staged to LDS as f16 (padded rows). Acc tile staged through LDS so the
// elementwise epilogue runs with coalesced b128 global traffic.
// ---------------------------------------------------------------------------
__global__ __launch_bounds__(256) void sgc_linear_wmma_kernel(
    const float* __restrict__ h,      // [N,64] after K hops
    const float* __restrict__ x,      // [N,64]
    const float* __restrict__ x_pre,  // [N,64]
    const float* __restrict__ W,      // [64,64], W[co][ci]
    const float* __restrict__ bias,   // [64]
    float* __restrict__ out,          // [N,64]
    int n_nodes) {
    __shared__ _Float16 Wh[CHN * WROW];        // ~9 KB, W in f16, padded rows
    __shared__ float    tile[8][16 * TROW];    // ~34 KB, per-wave 16x64 acc tile

    int tid = threadIdx.x;
    // Cooperative stage of W -> LDS (f32 -> f16), 1024 float4 groups / 256 thr.
    for (int g = tid; g < (CHN * CHN) / 4; g += 256) {
        float4 w4 = ((const float4*)W)[g];
        int row = g >> 4;
        int col = (g & 15) * 4;
        _Float16* p = &Wh[row * WROW + col];
        p[0] = (_Float16)w4.x;
        p[1] = (_Float16)w4.y;
        p[2] = (_Float16)w4.z;
        p[3] = (_Float16)w4.w;
    }
    __syncthreads();

    int wave = tid >> 5;
    int lane = tid & 31;
    int row_block = blockIdx.x * 8 + wave;       // 16 nodes per row block
    if (row_block * 16 >= n_nodes) return;       // whole-wave exit: EXEC all-1s for WMMA
    int n0 = row_block * 16;
    int m  = lane & 15;                          // tile row (A) / tile col (B,C,D)
    int hi = lane >> 4;                          // half-wave select
    float* mytile = &tile[wave][0];

    // --- A tiles (16x32 f16 each), documented 16-bit A layout:
    //   lane L: M = L&15; elems 0..7 -> K = 8*(L>>4)+i ; elems 8..15 -> +16
    const float* hrow = h + (size_t)(n0 + m) * CHN;
    v16h A[2];
#pragma unroll
    for (int kk = 0; kk < 2; ++kk) {
        int kb = kk * 32 + hi * 8;
#pragma unroll
        for (int i = 0; i < 8; ++i) {
            A[kk][i]     = (_Float16)hrow[kb + i];
            A[kk][8 + i] = (_Float16)hrow[kb + 16 + i];
        }
    }

    // --- 4 output channel tiles; park each result in LDS
#pragma unroll
    for (int co0 = 0; co0 < CHN; co0 += 16) {
        v8f acc = {};
#pragma unroll
        for (int kk = 0; kk < 2; ++kk) {
            // B (32x16 f16): lane holds col = m; elems 0..7 -> K = 8*hi + i, elems 8..15 -> +16 (+32*kk)
            // B[k][col] = W[co0+col][k] -> two 8-half runs of one padded W row
            const v8h* wp = (const v8h*)&Wh[(co0 + m) * WROW + kk * 32 + hi * 8];
            v8h b0 = wp[0];
            v8h b1 = wp[2];
            v16h B;
#pragma unroll
            for (int i = 0; i < 8; ++i) { B[i] = b0[i]; B[8 + i] = b1[i]; }
            acc = __builtin_amdgcn_wmma_f32_16x16x32_f16(
                false, A[kk], false, B, (short)0, acc, false, false);
        }
        // D layout: VGPR r -> row (r + 8*hi), col = m. Conflict-free scatter to LDS.
#pragma unroll
        for (int r = 0; r < 8; ++r)
            mytile[(r + 8 * hi) * TROW + co0 + m] = acc[r];
    }

    // --- coalesced epilogue: 16 rows x 64 cols = 256 float4 groups / 32 lanes
#pragma unroll
    for (int j = 0; j < 8; ++j) {
        int g   = j * 32 + lane;
        int row = g >> 4;
        int c4  = (g & 15) * 4;
        size_t gidx = (size_t)(n0 + row) * CHN + c4;
        float4 tv = *(const float4*)&mytile[row * TROW + c4];
        float4 xv = *(const float4*)(x + gidx);
        float4 xp = *(const float4*)(x_pre + gidx);
        float4 bv = *(const float4*)(bias + c4);
        float4 o;
        o.x = 2.0f * xv.x + DTDT * (tv.x + bv.x) - xp.x;
        o.y = 2.0f * xv.y + DTDT * (tv.y + bv.y) - xp.y;
        o.z = 2.0f * xv.z + DTDT * (tv.z + bv.z) - xp.z;
        o.w = 2.0f * xv.w + DTDT * (tv.w + bv.w) - xp.w;
        *(float4*)(out + gidx) = o;
    }
}

// ---------------------------------------------------------------------------
static inline size_t align_up(size_t v, size_t a) { return (v + a - 1) & ~(a - 1); }

extern "C" void kernel_launch(void* const* d_in, const int* in_sizes, int n_in,
                              void* d_out, int out_size, void* d_ws, size_t ws_size,
                              hipStream_t stream) {
    const float* x     = (const float*)d_in[0];   // [N,64]
    const float* x_pre = (const float*)d_in[1];   // [N,64]
    const int*   eidx  = (const int*)d_in[2];     // [2,E]
    const float* W     = (const float*)d_in[3];   // [64,64]
    const float* b     = (const float*)d_in[4];   // [64]
    float* out = (float*)d_out;

    const int N = in_sizes[0] / CHN;
    const int E = in_sizes[2] / 2;
    const int* src = eidx;        // edge_index[0]
    const int* dst = eidx + E;    // edge_index[1]

    // workspace carve-up
    char* ws = (char*)d_ws;
    size_t off = 0;
    float* dinv = (float*)(ws + off);                 off = align_up(off + (size_t)N * 4, 4096);
    float* hA   = (float*)(ws + off);                 off = align_up(off + (size_t)N * CHN * 4, 4096);
    float* hB   = (float*)(ws + off);
    (void)ws_size;

    const int TB = 256;

    // 1) degrees & symmetric norm
    deg_init_kernel<<<(N + TB - 1) / TB, TB, 0, stream>>>(dinv, N);
    deg_count_kernel<<<(E + TB - 1) / TB, TB, 0, stream>>>(dst, dinv, E);
    dinv_kernel<<<(N + TB - 1) / TB, TB, 0, stream>>>(dinv, N);

    // 2) K=4 hops of propagation: x -> hA -> hB -> hA -> hB
    const int ngroups = N * (CHN / 4);
    const int init_blocks = (ngroups + TB - 1) / TB;
    const int scat_blocks = (int)(((size_t)E * 32 + TB - 1) / TB);

    const float* cur = x;
    float* nxt = hA;
    for (int hop = 0; hop < 4; ++hop) {
        hop_init_kernel<<<init_blocks, TB, 0, stream>>>(cur, dinv, nxt, ngroups);
        edge_scatter_kernel<<<scat_blocks, TB, 0, stream>>>(cur, dinv, src, dst, nxt, E);
        cur = nxt;
        nxt = (nxt == hA) ? hB : hA;
    }

    // 3) fused SGConv linear + wave-poly combine via WMMA
    const int row_blocks = (N + 15) / 16;
    const int gemm_blocks = (row_blocks + 7) / 8;
    sgc_linear_wmma_kernel<<<gemm_blocks, TB, 0, stream>>>(
        cur, x, x_pre, W, b, out, N);
}
